// ConvDecoder_78563541779053
// MI455X (gfx1250) — matrix operand order repaired and checked by reading, not compiled, not run on hardware
//
#include <hip/hip_runtime.h>
#include <hip/hip_bf16.h>
#include <math.h>

// ---- problem constants -----------------------------------------------------
#define B_    32
#define T_    96
#define HID_  256
#define CE_   400
#define CEP_  416          // CE padded to multiple of 32
#define V_    256
#define HW_   1024         // 16 x 64
#define A_    256
#define MROWS (B_ * T_)    // 3072
#define PR_   22           // padded coverage rows (16 + 2 halo + extra for K-pad taps)
#define PC_   68           // padded coverage cols (64 + 2 + 2)
#define PIMG  (PR_ * PC_)  // 1496

typedef __attribute__((ext_vector_type(16))) _Float16 v16h;
typedef __attribute__((ext_vector_type(8)))  float    v8f;

union HReg { v16h h; _Float16 e[16]; };

// One 16x16(M,N) f32 tile accumulation over a K=32 slab via
// V_WMMA_F32_16X16X32_F16. Per-lane packing per the CDNA5 ISA layouts:
//   A 16x32: lane(l,half) row M=l; e[0..7]=K(half*8+i), e[8..15]=K(16+half*8+i)
//   B 32x16: lane(l,half) col N=l; e[0..7]=K(half*8+i), e[8..15]=K(16+half*8+i)
//   C/D    : lane(l,half) col N=l; VGPR i -> row M = half*8 + i
// NOTE: fa is always invoked with m==l and fb with n==l, so callers may
// pre-bias pointers by l.
template <typename FA, typename FB>
__device__ inline v8f wmma_tile_step(v8f c, FA fa, FB fb) {
  const int lane = threadIdx.x & 31;
  const int half = lane >> 4;
  const int l    = lane & 15;
  HReg a, b;
#pragma unroll
  for (int i = 0; i < 8; ++i) {
    a.e[i]     = fa(l, half * 8 + i);
    a.e[8 + i] = fa(l, 16 + half * 8 + i);
  }
#pragma unroll
  for (int i = 0; i < 8; ++i) {
    b.e[i]     = fb(half * 8 + i, l);
    b.e[8 + i] = fb(16 + half * 8 + i, l);
  }
  return __builtin_amdgcn_wmma_f32_16x16x32_f16(false, a.h, false, b.h,
                                                (short)0, c, false, false);
}

// ---- packing / utility kernels ---------------------------------------------
__global__ void zero_f32(float* __restrict__ p, int n) {
  int i = blockIdx.x * 256 + threadIdx.x;
  if (i < n) p[i] = 0.f;
}
__global__ void zero_f16(_Float16* __restrict__ p, int n) {
  int i = blockIdx.x * 256 + threadIdx.x;
  if (i < n) p[i] = (_Float16)0.f;
}

__global__ void embed_gather(const float* __restrict__ embed,
                             const int* __restrict__ tgt,
                             _Float16* __restrict__ X0) {
  int idx = blockIdx.x * 256 + threadIdx.x;   // MROWS*HID
  int r = idx >> 8, c = idx & 255;
  X0[idx] = (_Float16)embed[tgt[r] * HID_ + c];
}

// w (512,256,3) f32 -> wt [k][oc][ic] f16
__global__ void pack_glu_w(const float* __restrict__ w, _Float16* __restrict__ wt) {
  int idx = blockIdx.x * 256 + threadIdx.x;   // 3*512*256
  int k = idx >> 17, rem = idx & 131071;
  int oc = rem >> 8, ic = rem & 255;
  wt[idx] = (_Float16)w[(oc * HID_ + ic) * 3 + k];
}

// generic: dst[n][kpad] = k<K ? src[n*lds + koff + k] : 0   (f32 -> f16)
__global__ void pack_rowpad(const float* __restrict__ src, int lds, int koff,
                            int K, int kpad, _Float16* __restrict__ dst, int total) {
  int idx = blockIdx.x * 256 + threadIdx.x;
  if (idx >= total) return;
  int n = idx / kpad, k = idx - n * kpad;
  dst[idx] = (k < K) ? (_Float16)src[n * lds + koff + k] : (_Float16)0.f;
}

// enc_feat (B,CE,HW) f32 -> efT (B,HW,CEP) f16, zero-padded K
__global__ void pack_ef(const float* __restrict__ ef, _Float16* __restrict__ efT) {
  long idx = (long)blockIdx.x * 256 + threadIdx.x;  // B*HW*CEP
  int c = (int)(idx % CEP_);
  long rem = idx / CEP_;
  int hw = (int)(rem & 1023);
  int b = (int)(rem >> 10);
  efT[idx] = (c < CE_) ? (_Float16)ef[((size_t)b * CE_ + c) * HW_ + hw]
                       : (_Float16)0.f;
}

__global__ void glu_act(const float* __restrict__ Y,
                        const float* __restrict__ bias,
                        _Float16* __restrict__ Xo) {
  int idx = blockIdx.x * 256 + threadIdx.x;   // MROWS*HID
  int r = idx >> 8, c = idx & 255;
  float av = Y[r * 512 + c] + bias[c];
  float gv = Y[r * 512 + 256 + c] + bias[256 + c];
  Xo[idx] = (_Float16)(av * (1.f / (1.f + __expf(-gv))));
}

// ---- causal GLU conv as GEMM: Y(3072x512) = sum_k Xshift_k @ Wk ------------
__global__ void __launch_bounds__(256)
glu_gemm(const _Float16* __restrict__ X,    // (3072,256) f16
         const _Float16* __restrict__ wt,   // [3][512][256] f16
         float* __restrict__ Y) {
  int wave = threadIdx.x >> 5;
  int tile = blockIdx.x * 8 + wave;   // 192 Mt * 32 Nt
  int nt = tile & 31, mt = tile >> 5;
  int r0 = mt * 16, n0 = nt * 16;
  const int lane = threadIdx.x & 31, half = lane >> 4, l = lane & 15;
  const int row = r0 + l;
  const int tmod = row % T_;
  v8f c = {};
  for (int k = 0; k < 3; ++k) {
    int shift = k - 2;                       // y[t] = sum_k x[t+k-2] w[k]
    bool valid = (tmod + shift) >= 0;
    const _Float16* ap = X + (size_t)(valid ? row + shift : row) * HID_;
    const _Float16 am = valid ? (_Float16)1.f : (_Float16)0.f;
    const _Float16* bp = wt + ((size_t)k * 512 + n0 + l) * HID_;
    for (int kk = 0; kk < HID_; kk += 32) {
      c = wmma_tile_step(c,
        [&](int, int kl) -> _Float16 { return ap[kk + kl] * am; },
        [&](int kl, int) -> _Float16 { return bp[kk + kl]; });
    }
  }
  int col = n0 + l;
#pragma unroll
  for (int i = 0; i < 8; ++i)
    Y[(r0 + half * 8 + i) * 512 + col] = c[i];
}

// ---- enc_proj (f16 out): per-b (A x HW) = W_enc @ enc_feat[b] + b_enc ------
__global__ void __launch_bounds__(256)
encproj_gemm(const _Float16* __restrict__ wenc,   // (A, CEP) f16, K-padded
             const _Float16* __restrict__ efT,    // (B, HW, CEP) f16
             const float* __restrict__ b_enc,
             _Float16* __restrict__ encp) {       // (B, A, HW) f16
  int wave = threadIdx.x >> 5;
  int tile = blockIdx.x * 8 + wave;   // 32 b * (16 Mt * 64 Nt)
  int b = tile >> 10, local = tile & 1023;
  int mt = local >> 6, nt = local & 63;
  int a0 = mt * 16, n0 = nt * 16;
  const int lane = threadIdx.x & 31, half = lane >> 4, l = lane & 15;
  const _Float16* ap = wenc + (size_t)(a0 + l) * CEP_;
  const _Float16* bp = efT + ((size_t)b * HW_ + n0 + l) * CEP_;
  v8f c = {};
  for (int kk = 0; kk < CEP_; kk += 32) {
    c = wmma_tile_step(c,
      [&](int, int kl) -> _Float16 { return ap[kk + kl]; },
      [&](int kl, int) -> _Float16 { return bp[kk + kl]; });
  }
  int col = n0 + l;
  _Float16* outb = encp + (size_t)b * A_ * HW_;
#pragma unroll
  for (int i = 0; i < 8; ++i) {
    int a = a0 + half * 8 + i;
    outb[a * HW_ + col] = (_Float16)(c[i] + b_enc[a]);
  }
}

// ---- Out(3072x256) = X(3072x256) @ W16(n,k)^T (+bias) ----------------------
__global__ void __launch_bounds__(256)
proj_gemm(const _Float16* __restrict__ X,     // (3072,256) f16
          const _Float16* __restrict__ W16,   // (256,256) f16 [n][k]
          const float* __restrict__ bias,
          float* __restrict__ Out) {
  int wave = threadIdx.x >> 5;
  int tile = blockIdx.x * 8 + wave;   // 192 Mt * 16 Nt
  int nt = tile & 15, mt = tile >> 4;
  int r0 = mt * 16, n0 = nt * 16;
  const int lane = threadIdx.x & 31, half = lane >> 4, l = lane & 15;
  const _Float16* ap = X + (size_t)(r0 + l) * HID_;
  const _Float16* bp = W16 + (size_t)(n0 + l) * HID_;
  v8f c = {};
  for (int kk = 0; kk < HID_; kk += 32) {
    c = wmma_tile_step(c,
      [&](int, int kl) -> _Float16 { return ap[kk + kl]; },
      [&](int kl, int) -> _Float16 { return bp[kk + kl]; });
  }
  int col = n0 + l;
  float bv = bias ? bias[col] : 0.f;
#pragma unroll
  for (int i = 0; i < 8; ++i)
    Out[(r0 + half * 8 + i) * 256 + col] = c[i] + bv;
}

// ---- logits = ctx_all @ fcWc^T + Lh + fc_b ---------------------------------
__global__ void __launch_bounds__(256)
lc_gemm(const _Float16* __restrict__ CTX,    // (3072, CEP) f16, pad zeroed
        const _Float16* __restrict__ fcWc,   // (256, CEP) f16, pad zeroed
        const float* __restrict__ fcb,
        const float* __restrict__ Lh,
        float* __restrict__ Out) {
  int wave = threadIdx.x >> 5;
  int tile = blockIdx.x * 8 + wave;   // 192 Mt * 16 Nt
  int nt = tile & 15, mt = tile >> 4;
  int r0 = mt * 16, n0 = nt * 16;
  const int lane = threadIdx.x & 31, half = lane >> 4, l = lane & 15;
  const _Float16* ap = CTX + (size_t)(r0 + l) * CEP_;
  const _Float16* bp = fcWc + (size_t)(n0 + l) * CEP_;
  v8f c = {};
  for (int kk = 0; kk < CEP_; kk += 32) {
    c = wmma_tile_step(c,
      [&](int, int kl) -> _Float16 { return ap[kk + kl]; },
      [&](int kl, int) -> _Float16 { return bp[kk + kl]; });
  }
  int col = n0 + l;
  float bv = fcb[col];
#pragma unroll
  for (int i = 0; i < 8; ++i) {
    int row = r0 + half * 8 + i;
    Out[row * V_ + col] = c[i] + Lh[row * V_ + col] + bv;
  }
}

// ---- scan part 1: coverage conv (one WMMA, K=25 padded->32) fused with
//      tanh(enc_proj + q + b_cov + covfeat) * v_attn reduced over A ----------
__global__ void __launch_bounds__(512)
attn_e(const float* __restrict__ covp_all,   // (B, PR, PC) f32 zero-halo image
       const _Float16* __restrict__ wcov16,  // (A, 32) f16, taps 25..31 zero
       const float* __restrict__ b_cov, const float* __restrict__ v_attn,
       const _Float16* __restrict__ encp,    // (B, A, HW) f16
       const float* __restrict__ Q,          // (3072, A) f32
       float* __restrict__ e_out, int t) {
  __shared__ float part[16][16];
  int wave = threadIdx.x >> 5;          // 16 waves: one A-tile each
  int b  = blockIdx.x >> 6;             // 32 b * 64 position tiles
  int mt = blockIdx.x & 63;
  int p0 = mt * 16;
  int n0 = wave * 16;
  const int lane = threadIdx.x & 31, half = lane >> 4, l = lane & 15;
  // A side: 5x5 neighborhood of padded coverage image; always in bounds,
  // taps >=25 read junk rows but are annihilated by zero B-weights.
  const int p = p0 + l;
  const float* covp = covp_all + (size_t)b * PIMG + (p >> 6) * PC_ + (p & 63);
  const _Float16* bp = wcov16 + (size_t)(n0 + l) * 32;
  v8f c = {};
  c = wmma_tile_step(c,
    [&](int, int kl) -> _Float16 {
      return (_Float16)covp[(kl / 5) * PC_ + (kl % 5)];
    },
    [&](int kl, int) -> _Float16 { return bp[kl]; });
  int a = n0 + l;
  float qv = Q[(b * T_ + t) * A_ + a] + b_cov[a];
  float vv = v_attn[a];
  const _Float16* ep = encp + ((size_t)b * A_ + a) * HW_;
  float s[8];
#pragma unroll
  for (int i = 0; i < 8; ++i) {
    int pp = p0 + half * 8 + i;
    s[i] = vv * tanhf(c[i] + (float)ep[pp] + qv);
  }
#pragma unroll
  for (int i = 0; i < 8; ++i) {
#pragma unroll
    for (int off = 1; off < 16; off <<= 1)
      s[i] += __shfl_xor(s[i], off, 32);
  }
  if (l == 0) {
#pragma unroll
    for (int i = 0; i < 8; ++i) part[wave][half * 8 + i] = s[i];
  }
  __syncthreads();
  if (threadIdx.x < 16) {
    float acc = 0.f;
#pragma unroll
    for (int w = 0; w < 16; ++w) acc += part[w][threadIdx.x];
    e_out[b * HW_ + p0 + threadIdx.x] = acc;
  }
}

// ---- scan part 2: softmax over HW, cov += alpha, ctx = enc_feat @ alpha ----
__global__ void softmax_ctx(const float* __restrict__ e_in,
                            const float* __restrict__ enc_feat,
                            float* __restrict__ covp_all,
                            _Float16* __restrict__ ctx_all, int t) {
  __shared__ float red[256];
  __shared__ float alpha[HW_];
  int b = blockIdx.x, tid = threadIdx.x;
  const float* eb = e_in + b * HW_;
  float mx = -1e30f;
  for (int i = tid; i < HW_; i += 256) mx = fmaxf(mx, eb[i]);
  red[tid] = mx; __syncthreads();
  for (int s2 = 128; s2 > 0; s2 >>= 1) {
    if (tid < s2) red[tid] = fmaxf(red[tid], red[tid + s2]);
    __syncthreads();
  }
  mx = red[0]; __syncthreads();
  float sum = 0.f;
  for (int i = tid; i < HW_; i += 256) {
    float v = __expf(eb[i] - mx);
    alpha[i] = v; sum += v;
  }
  red[tid] = sum; __syncthreads();
  for (int s2 = 128; s2 > 0; s2 >>= 1) {
    if (tid < s2) red[tid] += red[tid + s2];
    __syncthreads();
  }
  float inv = 1.f / red[0];
  float* covp = covp_all + (size_t)b * PIMG;
  for (int i = tid; i < HW_; i += 256) {
    float av = alpha[i] * inv;
    alpha[i] = av;
    covp[(2 + (i >> 6)) * PC_ + 2 + (i & 63)] += av;   // interior of padded img
  }
  __syncthreads();
  const float* ef = enc_feat + (size_t)b * CE_ * HW_;
  _Float16* ctx = ctx_all + ((size_t)b * T_ + t) * CEP_;
  for (int ch = tid; ch < CE_; ch += 256) {
    const float* row = ef + ch * HW_;
    float acc = 0.f;
    for (int i = 0; i < HW_; ++i) acc += row[i] * alpha[i];
    ctx[ch] = (_Float16)acc;
  }
}

// ---- orchestration ---------------------------------------------------------
extern "C" void kernel_launch(void* const* d_in, const int* in_sizes, int n_in,
                              void* d_out, int out_size, void* d_ws, size_t ws_size,
                              hipStream_t stream) {
  const float* enc_feat = (const float*)d_in[0];
  const int*   tgt_in   = (const int*)d_in[1];
  const float* embed    = (const float*)d_in[2];
  const float* w1 = (const float*)d_in[3];
  const float* b1 = (const float*)d_in[4];
  const float* w2 = (const float*)d_in[5];
  const float* b2 = (const float*)d_in[6];
  const float* w3 = (const float*)d_in[7];
  const float* b3 = (const float*)d_in[8];
  const float* W_enc  = (const float*)d_in[9];
  const float* b_enc  = (const float*)d_in[10];
  const float* W_h    = (const float*)d_in[11];
  const float* b_h    = (const float*)d_in[12];
  const float* W_cov  = (const float*)d_in[13];
  const float* b_cov  = (const float*)d_in[14];
  const float* v_attn = (const float*)d_in[15];
  const float* fc_W   = (const float*)d_in[16];
  const float* fc_b   = (const float*)d_in[17];
  float* out = (float*)d_out;

  char* base = (char*)d_ws;
  size_t off = 0;
  auto alloc = [&](size_t bytes) -> void* {
    void* p = base + off;
    off = (off + bytes + 255) & ~(size_t)255;
    return p;
  };
  _Float16* X0     = (_Float16*)alloc((size_t)MROWS * HID_ * 2);
  _Float16* X1     = (_Float16*)alloc((size_t)MROWS * HID_ * 2);
  _Float16* X2     = (_Float16*)alloc((size_t)MROWS * HID_ * 2);
  _Float16* X3     = (_Float16*)alloc((size_t)MROWS * HID_ * 2);   // hidd
  float*    Ybuf   = (float*)   alloc((size_t)MROWS * 512 * 4);
  float*    Qall   = (float*)   alloc((size_t)MROWS * A_ * 4);
  float*    Lh     = (float*)   alloc((size_t)MROWS * V_ * 4);
  _Float16* encp   = (_Float16*)alloc((size_t)B_ * A_ * HW_ * 2);
  float*    covp   = (float*)   alloc((size_t)B_ * PIMG * 4);
  float*    ebuf   = (float*)   alloc((size_t)B_ * HW_ * 4);
  _Float16* ctx16  = (_Float16*)alloc((size_t)MROWS * CEP_ * 2);
  _Float16* wglu16 = (_Float16*)alloc((size_t)3 * 3 * 512 * HID_ * 2);
  _Float16* wenc16 = (_Float16*)alloc((size_t)A_ * CEP_ * 2);
  _Float16* efT16  = (_Float16*)alloc((size_t)B_ * HW_ * CEP_ * 2);
  _Float16* wh16   = (_Float16*)alloc((size_t)A_ * HID_ * 2);
  _Float16* fcWh16 = (_Float16*)alloc((size_t)V_ * HID_ * 2);
  _Float16* fcWc16 = (_Float16*)alloc((size_t)V_ * CEP_ * 2);
  _Float16* wcov16 = (_Float16*)alloc((size_t)A_ * 32 * 2);

  // ---- pack weights / operands to padded f16 ----
  pack_glu_w<<<1536, 256, 0, stream>>>(w1, wglu16);
  pack_glu_w<<<1536, 256, 0, stream>>>(w2, wglu16 + 3 * 512 * HID_);
  pack_glu_w<<<1536, 256, 0, stream>>>(w3, wglu16 + 6 * 512 * HID_);
  pack_rowpad<<<(A_ * CEP_ + 255) / 256, 256, 0, stream>>>(W_enc, CE_, 0, CE_, CEP_, wenc16, A_ * CEP_);
  pack_rowpad<<<(A_ * HID_ + 255) / 256, 256, 0, stream>>>(W_h, HID_, 0, HID_, HID_, wh16, A_ * HID_);
  pack_rowpad<<<(V_ * HID_ + 255) / 256, 256, 0, stream>>>(fc_W, HID_ + CE_, 0, HID_, HID_, fcWh16, V_ * HID_);
  pack_rowpad<<<(V_ * CEP_ + 255) / 256, 256, 0, stream>>>(fc_W, HID_ + CE_, HID_, CE_, CEP_, fcWc16, V_ * CEP_);
  pack_rowpad<<<(A_ * 32 + 255) / 256, 256, 0, stream>>>(W_cov, 25, 0, 25, 32, wcov16, A_ * 32);
  pack_ef<<<(int)(((size_t)B_ * HW_ * CEP_ + 255) / 256), 256, 0, stream>>>(enc_feat, efT16);

  zero_f32<<<(B_ * PIMG + 255) / 256, 256, 0, stream>>>(covp, B_ * PIMG);
  zero_f16<<<(MROWS * CEP_ + 255) / 256, 256, 0, stream>>>(ctx16, MROWS * CEP_);

  // ---- decoder trunk ----
  embed_gather<<<MROWS, 256, 0, stream>>>(embed, tgt_in, X0);
  glu_gemm<<<768, 256, 0, stream>>>(X0, wglu16, Ybuf);
  glu_act <<<MROWS, 256, 0, stream>>>(Ybuf, b1, X1);
  glu_gemm<<<768, 256, 0, stream>>>(X1, wglu16 + 3 * 512 * HID_, Ybuf);
  glu_act <<<MROWS, 256, 0, stream>>>(Ybuf, b2, X2);
  glu_gemm<<<768, 256, 0, stream>>>(X2, wglu16 + 6 * 512 * HID_, Ybuf);
  glu_act <<<MROWS, 256, 0, stream>>>(Ybuf, b3, X3);

  // ---- scan-invariant GEMMs ----
  encproj_gemm<<<4096, 256, 0, stream>>>(wenc16, efT16, b_enc, encp);
  proj_gemm<<<384, 256, 0, stream>>>(X3, wh16, b_h, Qall);        // Q_all
  proj_gemm<<<384, 256, 0, stream>>>(X3, fcWh16, nullptr, Lh);    // Lh

  // ---- sequential attention scan ----
  for (int t = 0; t < T_; ++t) {
    attn_e<<<B_ * 64, 512, 0, stream>>>(covp, wcov16, b_cov, v_attn, encp,
                                        Qall, ebuf, t);
    softmax_ctx<<<B_, 256, 0, stream>>>(ebuf, enc_feat, covp, ctx16, t);
  }

  // ---- final logits ----
  lc_gemm<<<384, 256, 0, stream>>>(ctx16, fcWc16, fc_b, Lh, out);
}
